// RNetAttention_49667001811785
// MI455X (gfx1250) — matrix operand order repaired and checked by reading, not compiled, not run on hardware
//
#include <hip/hip_runtime.h>
#include <hip/hip_bf16.h>

// ---------------------------------------------------------------------------
// R-Net attention + GRU scan for MI455X (gfx1250, wave32, WMMA bf16)
//
// Decomposition:
//   B=64 batch rows -> 4 groups x 16 rows. Each group = 16 workgroups
//   (one per WGP), each WG owns a 1/16 slice of every weight matrix's
//   OUTPUT dimension, held in LDS as pre-swizzled bf16 WMMA B-fragments
//   (256KB of the 320KB WGP LDS). The P=512 recurrence runs inside one
//   persistent kernel; the 16 WGs of a group sync with L2 atomic barriers
//   (4 per step). All dense layers: v_wmma_f32_16x16x32_bf16, M=16.
//   Weight slices are DMA'd into LDS with TENSOR_LOAD_TO_LDS (TDM,
//   TENSORcnt) when the builtin is available.
// ---------------------------------------------------------------------------

typedef unsigned short u16;
typedef unsigned int   u32;

typedef __attribute__((ext_vector_type(16))) __bf16 bf16x16;
typedef __attribute__((ext_vector_type(8)))  float  floatx8;
typedef __attribute__((ext_vector_type(4)))  u32    u32x4;
typedef __attribute__((ext_vector_type(4)))  int    i32x4;
typedef __attribute__((ext_vector_type(8)))  int    i32x8;

struct alignas(16) U128 { u32 a, b, c, d; };
struct alignas(32) U256 { U128 lo, hi; };

enum : int { Bb = 64, Pp = 512, Qq = 64, Dd = 512, Hh = 256,
             TWO_D = 1024, THREE_H = 768,
             NGROUP = 4, WPG = 16, MG = 16 };

// LDS carve (bytes). Dynamic LDS segment starts at offset 0 (no static
// __shared__ objects in k_main), so these offsets are also the raw LDS byte
// addresses used in the TDM descriptors.
enum : int {
  WV_OFF  = 0,        // 8   frags * 1024B =   8192
  WG_OFF  = 8192,     // 128 frags * 1024B = 131072
  WIH_OFF = 139264,   // 96  frags * 1024B =  98304
  WHH_OFF = 237568,   // 24  frags * 1024B =  24576
  VP_OFF  = 262144,   // 16x16 f32         =   1024
  GH_OFF  = 263168,   // 16x48 f32         =   3072
  GI_OFF  = 266240,   // 16x48 f32         =   3072
  AL_OFF  = 269312,   // 16x64 f32         =   4096
  SMEM_BYTES = 273408
};

#if defined(__HIP_DEVICE_COMPILE__) && __has_builtin(__builtin_amdgcn_tensor_load_to_lds)
#define USE_TDM 1
#else
#define USE_TDM 0
#endif

// ---------------------------------------------------------------------------
// helpers
// ---------------------------------------------------------------------------
__device__ __forceinline__ u16 f2bf(float f) {
  u32 u = __float_as_uint(f);
  u32 r = u + 0x7FFFu + ((u >> 16) & 1u);   // round-to-nearest-even
  return (u16)(r >> 16);
}

__device__ __forceinline__ float fast_sigmoid(float x) {
  return 1.0f / (1.0f + __expf(-x));
}

__device__ __forceinline__ float fast_tanh(float x) {
  x = fminf(fmaxf(x, -15.0f), 15.0f);
  float e = __expf(2.0f * x);
  return (e - 1.0f) / (e + 1.0f);
}

// A-fragment (16x32 bf16, M=lane&15, K-half=lane>>4) from a row-major bf16 row
// pointer already offset to (row m, col kbase). Layout: halfs [0..16) at
// K = kbase + 16*half + e (matches k_pack_b B order).
__device__ __forceinline__ bf16x16 load_a_frag(const u16* rowp, int hf) {
  U256 t;
  t.lo = *reinterpret_cast<const U128*>(rowp + hf * 16);
  t.hi = *reinterpret_cast<const U128*>(rowp + hf * 16 + 8);
  return __builtin_bit_cast(bf16x16, t);
}

// B-fragment from pre-swizzled buffer: 32B contiguous per lane.
__device__ __forceinline__ bf16x16 load_b_frag(const u16* fragp, int lane) {
  const U128* p = reinterpret_cast<const U128*>(fragp + lane * 16);
  U256 t; t.lo = p[0]; t.hi = p[1];
  return __builtin_bit_cast(bf16x16, t);
}

__device__ __forceinline__ floatx8 wmma_bf16(bf16x16 a, bf16x16 b, floatx8 c) {
  return __builtin_amdgcn_wmma_f32_16x16x32_bf16(
      false, a, false, b, (short)0, c, false, false);
}

__device__ __forceinline__ void cp_u128(U128* dst, const U128* src, int n) {
  for (int i = threadIdx.x; i < n; i += blockDim.x) dst[i] = src[i];
}

__device__ __forceinline__ void wait_tensorcnt0() {
#if __has_builtin(__builtin_amdgcn_s_wait_tensorcnt)
  __builtin_amdgcn_s_wait_tensorcnt(0);
#else
  asm volatile("s_wait_tensorcnt 0x0" ::: "memory");
#endif
}

#if USE_TDM
// 1-D TDM load: `words8` 8-byte elements from global -> LDS byte offset.
// D# group0: count=1 | lds_addr | global_addr[56:0] | type=2 ("image").
// D# group1: data_size=8B, tensor_dim0=tile_dim0=words8, dim1=1, stride=dim0.
// Issued once per wave (EXEC ignored by TDM); completion via TENSORcnt.
__device__ __forceinline__ void tdm_load_1d(const void* gsrc, u32 lds_off,
                                            u32 words8) {
  unsigned long long ga = (unsigned long long)(size_t)gsrc;
  u32x4 g0;
  g0[0] = 1u;                                           // count=1, user mode
  g0[1] = lds_off;                                      // lds_addr (bytes)
  g0[2] = (u32)(ga & 0xFFFFFFFFu);                      // global_addr[31:0]
  g0[3] = (u32)((ga >> 32) & 0x01FFFFFFu) | (2u << 30); // addr[56:32] | type=2
  i32x8 g1;
  g1[0] = (int)(3u << 16);                    // workgroup_mask=0, data_size=8B
  g1[1] = (int)((words8 & 0xFFFFu) << 16);    // tensor_dim0[15:0] @ bit48
  g1[2] = (int)((words8 >> 16) | (1u << 16)); // tensor_dim0[31:16] | dim1=1
  g1[3] = (int)(words8 << 16);                // dim1 hi=0 | tile_dim0
  g1[4] = 1;                                  // tile_dim1=1, tile_dim2=0
  g1[5] = (int)words8;                        // tensor_dim0_stride[31:0]
  g1[6] = 0;                                  // stride hi | dim1_stride lo
  g1[7] = 0;
  i32x4 z4 = {0, 0, 0, 0};
#if __clang_major__ >= 23
  i32x8 z8 = {0, 0, 0, 0, 0, 0, 0, 0};
  __builtin_amdgcn_tensor_load_to_lds(g0, g1, z4, z4, z8, 0);
#else
  __builtin_amdgcn_tensor_load_to_lds(g0, g1, z4, z4, 0);
#endif
}
#endif

// 16-WG group barrier (sense-reversing, agent scope). bar[0]=count, bar[1]=epoch.
__device__ __forceinline__ void group_barrier(u32* bar, u32* ep) {
  __syncthreads();
  if (threadIdx.x == 0) {
    __threadfence();
    u32 e = *ep;
    u32 prev = __hip_atomic_fetch_add(&bar[0], 1u, __ATOMIC_ACQ_REL,
                                      __HIP_MEMORY_SCOPE_AGENT);
    if (prev == (u32)(WPG - 1)) {
      __hip_atomic_store(&bar[0], 0u, __ATOMIC_RELAXED, __HIP_MEMORY_SCOPE_AGENT);
      __hip_atomic_store(&bar[1], e + 1u, __ATOMIC_RELEASE, __HIP_MEMORY_SCOPE_AGENT);
    } else {
      while (__hip_atomic_load(&bar[1], __ATOMIC_ACQUIRE,
                               __HIP_MEMORY_SCOPE_AGENT) == e)
        __builtin_amdgcn_s_sleep(1);
    }
  }
  __syncthreads();
  ++*ep;
}

// ---------------------------------------------------------------------------
// phase-0 kernels
// ---------------------------------------------------------------------------
__global__ void k_f32_to_bf16(const float* __restrict__ s, u16* __restrict__ d,
                              size_t n) {
  size_t stride = (size_t)gridDim.x * blockDim.x;
  for (size_t i = (size_t)blockIdx.x * blockDim.x + threadIdx.x; i < n; i += stride)
    d[i] = f2bf(s[i]);
}

// Pack torch weight W (Ntot x Ktot, row-major, y = x@W.T) into WMMA B-fragment
// order: frag (nt, kt), lane l, elem e:  N = nt*16 + (l&15),
// K = kt*32 + ((l>>4)<<4) + e. frag stride = 512 halfs = 1024B.
__global__ void k_pack_b(const float* __restrict__ W, u16* __restrict__ dst,
                         int Ntot, int Ktot) {
  int KT = Ktot >> 5;
  size_t total = (size_t)Ntot * (size_t)Ktot;
  size_t stride = (size_t)gridDim.x * blockDim.x;
  for (size_t i = (size_t)blockIdx.x * blockDim.x + threadIdx.x; i < total; i += stride) {
    size_t frag = i >> 9;
    int idx  = (int)(i & 511);
    int lane = idx >> 4, e = idx & 15;
    int nt = (int)(frag / KT), kt = (int)(frag % KT);
    int n = nt * 16 + (lane & 15);
    int k = kt * 32 + ((lane >> 4) << 4) + e;
    dst[i] = f2bf(W[(size_t)n * Ktot + k]);
  }
}

// Out(MxN,f32) = A(MxK,bf16 row-major) @ B(frags) + bias.  One wave per 16x16 tile.
__global__ __launch_bounds__(256) void k_gemm_bf16(
    const u16* __restrict__ A, const u16* __restrict__ Bf,
    const float* __restrict__ bias, float* __restrict__ Out,
    int M, int N, int K) {
  int wave = threadIdx.x >> 5, lane = threadIdx.x & 31;
  int tile = blockIdx.x * 8 + wave;
  int ntN = N >> 4;
  int mt = tile / ntN, nt = tile % ntN;
  if (mt * 16 >= M) return;
  int KT = K >> 5;
  int m16 = lane & 15, hf = lane >> 4;
  const u16* rowp  = A + (size_t)(mt * 16 + m16) * (size_t)K;
  const u16* bbase = Bf + (size_t)nt * KT * 512;
  floatx8 c = {0.f, 0.f, 0.f, 0.f, 0.f, 0.f, 0.f, 0.f};
  for (int kt = 0; kt < KT; ++kt) {
    bf16x16 a = load_a_frag(rowp + kt * 32, hf);
    bf16x16 b = load_b_frag(bbase + (size_t)kt * 512, lane);
    c = wmma_bf16(a, b, c);
  }
  int n = nt * 16 + m16;
  float bval = bias[n];
#pragma unroll
  for (int r = 0; r < 8; ++r)
    Out[(size_t)(mt * 16 + r + 8 * hf) * N + n] = c[r] + bval;
}

__global__ void k_zero32(u32* __restrict__ p, size_t n) {
  size_t stride = (size_t)gridDim.x * blockDim.x;
  for (size_t i = (size_t)blockIdx.x * blockDim.x + threadIdx.x; i < n; i += stride)
    p[i] = 0u;
}

// ---------------------------------------------------------------------------
// persistent scan kernel: 64 WGs = 4 groups x 16 WGs, 256 threads (8 waves)
// ---------------------------------------------------------------------------
__global__ __launch_bounds__(256) void k_main(
    const float* __restrict__ u_p, const float* __restrict__ u_q,
    const unsigned char* __restrict__ q_mask, const float* __restrict__ v_param,
    const float* __restrict__ bv, const float* __restrict__ bg,
    const float* __restrict__ b_ih, const float* __restrict__ b_hh,
    const u16* __restrict__ u_p_bf, const float* __restrict__ p_proj,
    const float* __restrict__ S_q,
    const u16* __restrict__ wvF, const u16* __restrict__ wgF,
    const u16* __restrict__ wihF, const u16* __restrict__ whhF,
    float* __restrict__ logitsPart, float* __restrict__ v_f32,
    u16* __restrict__ v_bf, float* __restrict__ c_f32, u16* __restrict__ c_bf,
    u16* __restrict__ gated_bf, u32* __restrict__ bars,
    float* __restrict__ out) {
  extern __shared__ char smem[];
  u16*   wvL  = (u16*)(smem + WV_OFF);
  u16*   wgL  = (u16*)(smem + WG_OFF);
  u16*   wihL = (u16*)(smem + WIH_OFF);
  u16*   whhL = (u16*)(smem + WHH_OFF);
  float* vpL  = (float*)(smem + VP_OFF);
  float* ghL  = (float*)(smem + GH_OFF);
  float* giL  = (float*)(smem + GI_OFF);
  float* aL   = (float*)(smem + AL_OFF);

  const int w  = blockIdx.x & 15;       // slice index within group
  const int g  = blockIdx.x >> 4;       // group index
  const int b0 = g * MG;                // global batch base
  const int tid  = threadIdx.x;
  const int wave = tid >> 5, lane = tid & 31;
  const int m16  = lane & 15, hf = lane >> 4;

  float* lpG = logitsPart + (size_t)g * (WPG * MG * Qq);
  float* vG  = v_f32 + (size_t)g * (MG * Hh);
  u16*   vbG = v_bf  + (size_t)g * (MG * Hh);
  float* cG  = c_f32 + (size_t)g * (MG * Dd);
  u16*   cbG = c_bf  + (size_t)g * (MG * Dd);
  u16*   gtG = gated_bf + (size_t)g * (MG * TWO_D);
  u32*   barG = bars + g * 16;

  // ---- preload this WG's weight slices into LDS ----
#if USE_TDM
  // One 1-D TDM descriptor per wave (8 DMAs, EXEC ignored, TENSORcnt-tracked):
  //  wave0: Wv (8KB)   wave1: Wg (128KB)
  //  waves 2-4: W_ih gate chunks (32KB each)  waves 5-7: W_hh chunks (8KB each)
  if (wave == 0) {
    tdm_load_1d(wvF + (size_t)w * 8 * 512, WV_OFF, (8 * 1024) / 8);
  } else if (wave == 1) {
    tdm_load_1d(wgF + (size_t)w * 128 * 512, WG_OFF, (128 * 1024) / 8);
  } else if (wave <= 4) {
    int t = wave - 2;
    tdm_load_1d(wihF + (size_t)(t * 16 + w) * 32 * 512,
                (u32)(WIH_OFF + t * 32 * 1024), (32 * 1024) / 8);
  } else {
    int t = wave - 5;
    tdm_load_1d(whhF + (size_t)(t * 16 + w) * 8 * 512,
                (u32)(WHH_OFF + t * 8 * 1024), (8 * 1024) / 8);
  }
  wait_tensorcnt0();
#else
  cp_u128((U128*)wvL, (const U128*)(wvF + (size_t)w * 8 * 512), 8 * 64);
  cp_u128((U128*)wgL, (const U128*)(wgF + (size_t)w * 128 * 512), 128 * 64);
#pragma unroll
  for (int t = 0; t < 3; ++t) {
    cp_u128((U128*)wihL + t * 32 * 64,
            (const U128*)(wihF + (size_t)(t * 16 + w) * 32 * 512), 32 * 64);
    cp_u128((U128*)whhL + t * 8 * 64,
            (const U128*)(whhF + (size_t)(t * 16 + w) * 8 * 512), 8 * 64);
  }
#endif
  __syncthreads();

  u32 ep = 0;

  for (int s = 0; s < Pp; ++s) {
    // ================= Stage A: vp slice, gh slice, score partials ========
    if (wave == 0) {
      // vp(16x16) = v(16x256) @ Wv.T slice + bv + p_proj[:, s, slice]
      floatx8 c = {0.f, 0.f, 0.f, 0.f, 0.f, 0.f, 0.f, 0.f};
      const u16* rowp = vbG + m16 * Hh;
#pragma unroll
      for (int kt = 0; kt < 8; ++kt) {
        bf16x16 a = load_a_frag(rowp + kt * 32, hf);
        bf16x16 b = load_b_frag(wvL + kt * 512, lane);
        c = wmma_bf16(a, b, c);
      }
      float bias = bv[w * 16 + m16];
#pragma unroll
      for (int r = 0; r < 8; ++r) {
        int bb = r + 8 * hf;
        vpL[bb * 16 + m16] =
            c[r] + bias +
            p_proj[(size_t)(b0 + bb) * Pp * Hh + (size_t)s * Hh + w * 16 + m16];
      }
    } else if (wave <= 3) {
      // gh(16x48) = v @ W_hh.T slice + b_hh   (gate tile t = wave-1)
      int t = wave - 1;
      floatx8 c = {0.f, 0.f, 0.f, 0.f, 0.f, 0.f, 0.f, 0.f};
      const u16* rowp = vbG + m16 * Hh;
#pragma unroll
      for (int kt = 0; kt < 8; ++kt) {
        bf16x16 a = load_a_frag(rowp + kt * 32, hf);
        bf16x16 b = load_b_frag(whhL + (size_t)(t * 8 + kt) * 512, lane);
        c = wmma_bf16(a, b, c);
      }
      float bias = b_hh[t * Hh + w * 16 + m16];
#pragma unroll
      for (int r = 0; r < 8; ++r)
        ghL[(r + 8 * hf) * 48 + t * 16 + m16] = c[r] + bias;
    } else if (wave == 4 && s + 1 < Pp && lane < 16) {
      // warm next step's p_proj rows
      __builtin_prefetch(
          p_proj + (size_t)(b0 + lane) * Pp * Hh + (size_t)(s + 1) * Hh + w * 16, 0, 2);
    }
    __syncthreads();

    // score partials over this WG's 16-wide h slice (deterministic slices)
#pragma unroll
    for (int i = 0; i < 4; ++i) {
      int pair = tid + 256 * i;           // 1024 (b,q) pairs
      int bb = pair >> 6, q = pair & 63;
      const float* sqp =
          S_q + (size_t)(b0 + bb) * Qq * Hh + (size_t)q * Hh + w * 16;
      float acc = 0.f;
#pragma unroll
      for (int hh = 0; hh < 16; ++hh)
        acc += fast_tanh(sqp[hh] + vpL[bb * 16 + hh]) * v_param[w * 16 + hh];
      lpG[(size_t)w * (MG * Qq) + pair] = acc;
    }
    group_barrier(barG, &ep);   // ---- B2: score partials visible ----

    // ================= Stage C: softmax + context slice ====================
#pragma unroll
    for (int i = 0; i < 4; ++i) {
      int pair = tid + 256 * i;
      float acc = 0.f;
#pragma unroll
      for (int ww = 0; ww < WPG; ++ww) acc += lpG[(size_t)ww * (MG * Qq) + pair];
      aL[pair] = acc;
    }
    __syncthreads();
    if (tid < MG) {   // per-row masked softmax (row bb)
      int bb = tid;
      float mx = -3.4e38f;
      for (int q = 0; q < Qq; ++q) {
        float x = q_mask[(b0 + bb) * Qq + q] ? aL[bb * Qq + q] : -1e30f;
        aL[bb * Qq + q] = x;
        mx = fmaxf(mx, x);
      }
      float sum = 0.f;
      for (int q = 0; q < Qq; ++q) {
        float e = __expf(aL[bb * Qq + q] - mx);
        aL[bb * Qq + q] = e;
        sum += e;
      }
      float inv = 1.0f / sum;
      for (int q = 0; q < Qq; ++q) aL[bb * Qq + q] *= inv;
    }
    __syncthreads();
    // c[:, 32w..32w+32) = a @ u_q (fp32 exact)
#pragma unroll
    for (int i = 0; i < 2; ++i) {
      int o = tid + 256 * i;              // 512 outputs
      int bb = o >> 5, dd = o & 31, d = w * 32 + dd;
      const float* uq = u_q + (size_t)(b0 + bb) * Qq * Dd + d;
      float acc = 0.f;
      for (int q = 0; q < Qq; ++q) acc += aL[bb * Qq + q] * uq[(size_t)q * Dd];
      cG[bb * Dd + d]  = acc;
      cbG[bb * Dd + d] = f2bf(acc);
    }
    group_barrier(barG, &ep);   // ---- B3: full inp = [c, up_i] ready ----

    // ================= Stage D: gated input slice ==========================
    if (wave < 4) {
      int jb = w * 64 + wave * 16;
      floatx8 c = {0.f, 0.f, 0.f, 0.f, 0.f, 0.f, 0.f, 0.f};
#pragma unroll 4
      for (int kt = 0; kt < 32; ++kt) {
        int kb = kt * 32;
        const u16* rowp =
            (kb < Dd)
                ? (cbG + m16 * Dd + kb)
                : (u_p_bf + (size_t)(b0 + m16) * Pp * Dd + (size_t)s * Dd + (kb - Dd));
        bf16x16 a = load_a_frag(rowp, hf);
        bf16x16 b = load_b_frag(wgL + (size_t)(wave * 32 + kt) * 512, lane);
        c = wmma_bf16(a, b, c);
      }
      int j = jb + m16;
      float bgj = bg[j];
#pragma unroll
      for (int r = 0; r < 8; ++r) {
        int bb = r + 8 * hf;
        float Gv = c[r] + bgj;
        float inpv = (j < Dd)
                         ? cG[bb * Dd + j]
                         : u_p[(size_t)(b0 + bb) * Pp * Dd + (size_t)s * Dd + (j - Dd)];
        gtG[bb * TWO_D + j] = f2bf(inpv * fast_sigmoid(Gv));
      }
    }
    group_barrier(barG, &ep);   // ---- B4: gated input ready ----

    // ================= Stage E: gi slice + GRU update ======================
    if (wave < 3) {
      int t = wave;
      floatx8 c = {0.f, 0.f, 0.f, 0.f, 0.f, 0.f, 0.f, 0.f};
      const u16* rowp = gtG + m16 * TWO_D;
#pragma unroll 4
      for (int kt = 0; kt < 32; ++kt) {
        bf16x16 a = load_a_frag(rowp + kt * 32, hf);
        bf16x16 b = load_b_frag(wihL + (size_t)(t * 32 + kt) * 512, lane);
        c = wmma_bf16(a, b, c);
      }
      float bias = b_ih[t * Hh + w * 16 + m16];
#pragma unroll
      for (int r = 0; r < 8; ++r)
        giL[(r + 8 * hf) * 48 + t * 16 + m16] = c[r] + bias;
    }
    __syncthreads();
    {
      int bb = tid >> 4, hh = tid & 15, h = w * 16 + hh;
      float gir = giL[bb * 48 + hh], giz = giL[bb * 48 + 16 + hh],
            gin = giL[bb * 48 + 32 + hh];
      float ghr = ghL[bb * 48 + hh], ghz = ghL[bb * 48 + 16 + hh],
            ghn = ghL[bb * 48 + 32 + hh];
      float r = fast_sigmoid(gir + ghr);
      float z = fast_sigmoid(giz + ghz);
      float n = fast_tanh(gin + r * ghn);
      float vo = vG[bb * Hh + h];
      float vn = (1.0f - z) * n + z * vo;
      vG[bb * Hh + h]  = vn;
      vbG[bb * Hh + h] = f2bf(vn);
      out[(size_t)(b0 + bb) * Pp * Hh + (size_t)s * Hh + h] = vn;
    }
    group_barrier(barG, &ep);   // ---- B5: v visible for next step ----
  }
}

// ---------------------------------------------------------------------------
// host launcher
// ---------------------------------------------------------------------------
extern "C" void kernel_launch(void* const* d_in, const int* in_sizes, int n_in,
                              void* d_out, int out_size, void* d_ws, size_t ws_size,
                              hipStream_t stream) {
  (void)in_sizes; (void)n_in; (void)out_size; (void)ws_size;
  const float* u_p     = (const float*)d_in[0];
  const float* u_q     = (const float*)d_in[1];
  const unsigned char* q_mask = (const unsigned char*)d_in[3];   // p_mask unused
  const float* Wp      = (const float*)d_in[4];
  const float* bp      = (const float*)d_in[5];
  const float* Wq      = (const float*)d_in[6];
  const float* bq      = (const float*)d_in[7];
  const float* Wv      = (const float*)d_in[8];
  const float* bv      = (const float*)d_in[9];
  const float* Wg      = (const float*)d_in[10];
  const float* bg      = (const float*)d_in[11];
  const float* W_ih    = (const float*)d_in[12];
  const float* W_hh    = (const float*)d_in[13];
  const float* b_ih    = (const float*)d_in[14];
  const float* b_hh    = (const float*)d_in[15];
  const float* v_param = (const float*)d_in[16];
  float* out = (float*)d_out;

  char* base = (char*)d_ws;
  size_t cur = 0;
  auto alloc = [&](size_t bytes) -> char* {
    cur = (cur + 255) & ~(size_t)255;
    char* p = base + cur;
    cur += bytes;
    return p;
  };

  u16*   upbf = (u16*)alloc((size_t)Bb * Pp * Dd * 2);
  u16*   uqbf = (u16*)alloc((size_t)Bb * Qq * Dd * 2);
  float* pprj = (float*)alloc((size_t)Bb * Pp * Hh * 4);
  float* sq   = (float*)alloc((size_t)Bb * Qq * Hh * 4);
  u16*   WpF  = (u16*)alloc((size_t)Hh * Dd * 2);
  u16*   WqF  = (u16*)alloc((size_t)Hh * Dd * 2);
  u16*   WvF  = (u16*)alloc((size_t)Hh * Hh * 2);
  u16*   WgF  = (u16*)alloc((size_t)TWO_D * TWO_D * 2);
  u16*   WihF = (u16*)alloc((size_t)THREE_H * TWO_D * 2);
  u16*   WhhF = (u16*)alloc((size_t)THREE_H * Hh * 2);

  size_t dynStart = (cur + 255) & ~(size_t)255;
  float* lp  = (float*)alloc((size_t)NGROUP * WPG * MG * Qq * 4);
  float* vf  = (float*)alloc((size_t)NGROUP * MG * Hh * 4);
  u16*   vb  = (u16*)alloc((size_t)NGROUP * MG * Hh * 2);
  float* cf  = (float*)alloc((size_t)NGROUP * MG * Dd * 4);
  u16*   cb  = (u16*)alloc((size_t)NGROUP * MG * Dd * 2);
  u16*   gt  = (u16*)alloc((size_t)NGROUP * MG * TWO_D * 2);
  u32*   bars = (u32*)alloc((size_t)NGROUP * 16 * 4);
  size_t dynEnd = (cur + 3) & ~(size_t)3;
  size_t dynWords = (dynEnd - dynStart) / 4;

  // phase 0: bf16 copies + B-fragment packing
  k_f32_to_bf16<<<4096, 256, 0, stream>>>(u_p, upbf, (size_t)Bb * Pp * Dd);
  k_f32_to_bf16<<<1024, 256, 0, stream>>>(u_q, uqbf, (size_t)Bb * Qq * Dd);
  k_pack_b<<<512, 256, 0, stream>>>(Wp, WpF, Hh, Dd);
  k_pack_b<<<512, 256, 0, stream>>>(Wq, WqF, Hh, Dd);
  k_pack_b<<<512, 256, 0, stream>>>(Wv, WvF, Hh, Hh);
  k_pack_b<<<1024, 256, 0, stream>>>(Wg, WgF, TWO_D, TWO_D);
  k_pack_b<<<1024, 256, 0, stream>>>(W_ih, WihF, THREE_H, TWO_D);
  k_pack_b<<<512, 256, 0, stream>>>(W_hh, WhhF, THREE_H, Hh);

  // phase 1: one-time WMMA GEMMs  p_proj = u_p@Wp.T+bp,  S_q = u_q@Wq.T+bq
  k_gemm_bf16<<<4096, 256, 0, stream>>>(upbf, WpF, bp, pprj, Bb * Pp, Hh, Dd);
  k_gemm_bf16<<<512, 256, 0, stream>>>(uqbf, WqF, bq, sq, Bb * Qq, Hh, Dd);

  // phase 2: zero dynamic state (v0 = 0, barriers) then run the scan
  k_zero32<<<256, 256, 0, stream>>>((u32*)(base + dynStart), dynWords);

  (void)hipFuncSetAttribute(reinterpret_cast<const void*>(k_main),
                            hipFuncAttributeMaxDynamicSharedMemorySize,
                            SMEM_BYTES);
  k_main<<<NGROUP * WPG, 256, SMEM_BYTES, stream>>>(
      u_p, u_q, q_mask, v_param, bv, bg, b_ih, b_hh,
      upbf, pprj, sq, WvF, WgF, WihF, WhhF,
      lp, vf, vb, cf, cb, gt, bars, out);
}
